// MultiHeadSelfAttention_25383256719766
// MI455X (gfx1250) — matrix operand order repaired and checked by reading, not compiled, not run on hardware
//
#include <hip/hip_runtime.h>
#include <hip/hip_bf16.h>

// ---------------------------------------------------------------------------
// MI455X (gfx1250) multi-head self-attention, bf16 WMMA pipeline.
//   x[2,2048,1024] fp32 -> bf16 -> Q/K/V proj (WMMA, async-LDS staged GEMM)
//   -> flash attention (WMMA) -> output proj -> fp32 out.
// ~69 GFLOP vs ~80MB HBM traffic -> matrix-core bound; all GEMM work via
// v_wmma_f32_16x16x32_bf16, operands staged with global_load_async_to_lds_b128
// (ASYNCcnt) and double-buffered in LDS.
// ---------------------------------------------------------------------------

#define DEV __device__ __forceinline__

typedef unsigned short u16;
typedef unsigned int   u32;
typedef __attribute__((ext_vector_type(16))) __bf16 bf16x16;
typedef __attribute__((ext_vector_type(8)))  float  f32x8;

constexpr int Bc  = 2;
constexpr int Sc  = 2048;
constexpr int Dc  = 1024;
constexpr int Hc  = 16;
constexpr int HDc = 64;
constexpr int Nc  = Bc * Sc;          // 4096 tokens
constexpr float SCALE = 0.125f;       // 1/sqrt(64)

DEV u16 f2bf(float f) {                       // round-to-nearest-even fp32->bf16
  u32 u = __builtin_bit_cast(u32, f);
  u += 0x7fffu + ((u >> 16) & 1u);
  return (u16)(u >> 16);
}

union FragU { bf16x16 v; uint4 q[2]; };

// Load a 32-byte bf16 fragment as two 16B chunks (global_load_b128 or
// ds_load_b128). second_off is in halfword units.
DEV bf16x16 ld_frag(const u16* p, int second_off) {
  FragU f;
  f.q[0] = *(const uint4*)(p);
  f.q[1] = *(const uint4*)(p + second_off);
  return f.v;
}

DEV f32x8 wmma_bf16(bf16x16 a, bf16x16 b, f32x8 c) {
  return __builtin_amdgcn_wmma_f32_16x16x32_bf16(
      /*neg_a=*/false, a, /*neg_b=*/false, b,
      /*c_mod=*/(short)0, c, /*reuse_a=*/false, /*reuse_b=*/false);
}

// Async copy 16B/lane global -> LDS (ASYNCcnt-tracked, CDNA5).
// Flat shared address truncates to the LDS byte address (aperture mapping).
DEV void async_cp16(const u16* lds_ptr, const u16* gptr) {
  u32 loff = (u32)(uintptr_t)lds_ptr;
  asm volatile("global_load_async_to_lds_b128 %0, %1, off"
               :: "v"(loff), "v"(gptr) : "memory");
}
DEV void wait_async0() { asm volatile("s_wait_asynccnt 0x0" ::: "memory"); }

// ---------------------------------------------------------------------------
// fp32 -> bf16 conversion, 4 elements / thread
// ---------------------------------------------------------------------------
__global__ void mhsa_cvt_bf16(const float* __restrict__ src,
                              u16* __restrict__ dst, int n) {
  int i4 = (blockIdx.x * blockDim.x + threadIdx.x) * 4;
  if (i4 < n) {
    float4 v = *(const float4*)(src + i4);
    ushort4 r;
    r.x = f2bf(v.x); r.y = f2bf(v.y); r.z = f2bf(v.z); r.w = f2bf(v.w);
    *(ushort4*)(dst + i4) = r;
  }
}

// ---------------------------------------------------------------------------
// Y = X (Nc x Dc, bf16) @ W^T (W is [out,in] bf16) + bias.
// Block tile 128x64, k-step 32, double-buffered async LDS staging.
// 8 waves: wave w computes rows [m0+16w, m0+16w+16) x 64 cols (4 C frags).
// mode 0: bf16 row-major [tok, o]  (Q, K)
// mode 1: bf16 per-head transpose [b,h,d,s]  (V)
// mode 2: fp32 row-major  (final projection)
// ---------------------------------------------------------------------------
__global__ void __launch_bounds__(256)
mhsa_gemm_bf16(const u16* __restrict__ X, const u16* __restrict__ W,
               const float* __restrict__ bias, void* __restrict__ out,
               int mode) {
  __shared__ __align__(16) u16 lx[2][128 * 32];   // 2 x 8KB  X tile
  __shared__ __align__(16) u16 lw[2][64 * 32];    // 2 x 4KB  W tile

  const int lane = threadIdx.x & 31;
  const int half = lane >> 4;
  const int l16  = lane & 15;
  const int w    = threadIdx.x >> 5;      // wave 0..7 -> m sub-tile
  const int mblk = blockIdx.x >> 4;       // 0..31  (4096/128)
  const int nblk = blockIdx.x & 15;       // 0..15  (1024/64)
  const int m0   = mblk * 128;
  const int n0   = nblk * 64;

  // Stage X(128x32) + W(64x32) tiles for k-step kk into buffer buf.
  // 512 + 256 16B transfers; 3 async instructions per wave.
  auto stage = [&](int buf, int kk) {
#pragma unroll
    for (int i = 0; i < 2; ++i) {
      const int id  = threadIdx.x + 256 * i;
      const int row = id >> 2, ch = (id & 3) * 8;
      async_cp16(&lx[buf][row * 32 + ch],
                 X + (size_t)(m0 + row) * Dc + kk + ch);
    }
    {
      const int row = threadIdx.x >> 2, ch = (threadIdx.x & 3) * 8;
      async_cp16(&lw[buf][row * 32 + ch],
                 W + (size_t)(n0 + row) * Dc + kk + ch);
    }
  };

  const f32x8 zf = {0.f,0.f,0.f,0.f,0.f,0.f,0.f,0.f};
  f32x8 acc[4] = {zf, zf, zf, zf};

  stage(0, 0);
  int buf = 0;
  for (int kk = 0; kk < Dc; kk += 32, buf ^= 1) {
    wait_async0();           // my tile contributions have landed in LDS
    __syncthreads();         // everyone's have
    if (kk + 32 < Dc) stage(buf ^ 1, kk + 32);   // overlap next stage

    // Batch all fragment loads (10x ds_load_b128), then a single s_wait_dscnt
    // and 4 back-to-back WMMAs (XDL ops co-execute with surrounding VALU/DS).
    bf16x16 a = ld_frag(&lx[buf][(w * 16 + l16) * 32] + half * 16, 8);
    bf16x16 bfr[4];
#pragma unroll
    for (int nt = 0; nt < 4; ++nt)
      bfr[nt] = ld_frag(&lw[buf][(nt * 16 + l16) * 32] + half * 16, 8);
#pragma unroll
    for (int nt = 0; nt < 4; ++nt)
      acc[nt] = wmma_bf16(a, bfr[nt], acc[nt]);

    __syncthreads();         // done reading buf before it is restaged
  }

  const int row0 = m0 + w * 16;
#pragma unroll
  for (int nt = 0; nt < 4; ++nt) {
    const int col = n0 + nt * 16 + l16;
    const float bv = bias[col];
#pragma unroll
    for (int j = 0; j < 8; ++j) {
      const int row = row0 + j + 8 * half;     // C layout: row = vgpr + 8*half
      const float val = acc[nt][j] + bv;
      if (mode == 2) {
        ((float*)out)[(size_t)row * Dc + col] = val;
      } else if (mode == 0) {
        ((u16*)out)[(size_t)row * Dc + col] = f2bf(val);
      } else {                                  // V transposed: [b,h,d,s]
        const int s = row & (Sc - 1);
        const int b = row >> 11;
        const int h = col >> 6;
        const int d = col & 63;
        ((u16*)out)[((size_t)(b * Hc + h) * HDc + d) * Sc + s] = f2bf(val);
      }
    }
  }
}

// ---------------------------------------------------------------------------
// Flash attention: one wave = 16 queries of one (b,h); streams 32-key tiles.
// Q,K bf16 [tok, D]; VT bf16 [b,h,d,s]; CTX bf16 [tok, D].
// ---------------------------------------------------------------------------
__global__ void __launch_bounds__(256)
mhsa_flash_attn(const u16* __restrict__ Q, const u16* __restrict__ K,
                const u16* __restrict__ VT, u16* __restrict__ CTX) {
  __shared__ __align__(16) u16 plds[8][16 * 32];   // P repack buffer per wave

  const int lane = threadIdx.x & 31;
  const int half = lane >> 4;
  const int l16  = lane & 15;
  const int w    = threadIdx.x >> 5;
  const int wid  = blockIdx.x * 8 + w;
  const int qtile = wid & 127;           // S/16 = 128 query tiles
  const int bh    = wid >> 7;            // 0..31
  const int b     = bh >> 4;
  const int h     = bh & 15;

  const int row0 = b * Sc + qtile * 16;  // global token row of this tile
  const u16* qrow = Q + (size_t)(row0 + l16) * Dc + h * HDc;
  const bf16x16 aQ0 = ld_frag(qrow + 0  + half * 16, 8);   // k = 0..31
  const bf16x16 aQ1 = ld_frag(qrow + 32 + half * 16, 8);   // k = 32..63

  const u16* kbase = K + (size_t)(b * Sc) * Dc + h * HDc;  // + key*Dc
  const u16* vbase = VT + (size_t)bh * HDc * Sc;           // + d*Sc + key
  u16* myp = &plds[w][0];

  const f32x8 zf = {0.f,0.f,0.f,0.f,0.f,0.f,0.f,0.f};
  f32x8 o[4] = {zf, zf, zf, zf};
  float m[8], l[8];
#pragma unroll
  for (int j = 0; j < 8; ++j) { m[j] = -1e30f; l[j] = 0.f; }

  for (int kt = 0; kt < Sc; kt += 32) {
    // ---- batch all K fragments, then 4 WMMAs (single wait, dual-issue) ----
    const u16* kc = kbase + (size_t)(kt + l16) * Dc + half * 16;
    bf16x16 b00 = ld_frag(kc, 8);
    bf16x16 b01 = ld_frag(kc + 32, 8);
    bf16x16 b10 = ld_frag(kc + 16 * Dc, 8);
    bf16x16 b11 = ld_frag(kc + 16 * Dc + 32, 8);
    // hoist V fragments: their latency hides under the softmax VALU work
    bf16x16 bV[4];
#pragma unroll
    for (int nt = 0; nt < 4; ++nt)
      bV[nt] = ld_frag(vbase + (size_t)(nt * 16 + l16) * Sc + kt + half * 16, 8);

    f32x8 s0 = zf, s1 = zf;
    s0 = wmma_bf16(aQ0, b00, s0);
    s0 = wmma_bf16(aQ1, b01, s0);
    s1 = wmma_bf16(aQ0, b10, s1);
    s1 = wmma_bf16(aQ1, b11, s1);

    // ---- online softmax (row reductions across the 16 lanes of a half) ----
#pragma unroll
    for (int j = 0; j < 8; ++j) {
      float a = s0[j] * SCALE;
      float c = s1[j] * SCALE;
      float mx = fmaxf(a, c);
      mx = fmaxf(mx, __shfl_xor(mx, 1, 32));
      mx = fmaxf(mx, __shfl_xor(mx, 2, 32));
      mx = fmaxf(mx, __shfl_xor(mx, 4, 32));
      mx = fmaxf(mx, __shfl_xor(mx, 8, 32));
      const float mn = fmaxf(m[j], mx);
      const float r  = __expf(m[j] - mn);
      m[j] = mn;
      const float p0 = __expf(a - mn);
      const float p1 = __expf(c - mn);
      float sm = p0 + p1;
      sm += __shfl_xor(sm, 1, 32);
      sm += __shfl_xor(sm, 2, 32);
      sm += __shfl_xor(sm, 4, 32);
      sm += __shfl_xor(sm, 8, 32);
      l[j] = l[j] * r + sm;
      o[0][j] *= r; o[1][j] *= r; o[2][j] *= r; o[3][j] *= r;
      // repack P (C layout -> row-major LDS) for the A-fragment of PV
      const int prow = (j + 8 * half) * 32;
      myp[prow + l16]      = f2bf(p0);
      myp[prow + 16 + l16] = f2bf(p1);
    }

    asm volatile("s_wait_dscnt 0x0" ::: "memory");  // P stores visible wave-wide

    // ---- O += P V  (A = P 16x32 from LDS; B = V^T columns, contiguous) ----
    const bf16x16 aP = ld_frag(myp + l16 * 32 + half * 16, 8);
#pragma unroll
    for (int nt = 0; nt < 4; ++nt)
      o[nt] = wmma_bf16(aP, bV[nt], o[nt]);
  }

  // ---- normalize and store context (bf16, [tok, D]) ----
#pragma unroll
  for (int j = 0; j < 8; ++j) {
    const float inv = 1.0f / l[j];
    const int row = row0 + j + 8 * half;
#pragma unroll
    for (int nt = 0; nt < 4; ++nt) {
      CTX[(size_t)row * Dc + h * HDc + nt * 16 + l16] = f2bf(o[nt][j] * inv);
    }
  }
}

// ---------------------------------------------------------------------------
extern "C" void kernel_launch(void* const* d_in, const int* in_sizes, int n_in,
                              void* d_out, int out_size, void* d_ws, size_t ws_size,
                              hipStream_t stream) {
  const float* x  = (const float*)d_in[0];
  const float* wq = (const float*)d_in[1];
  const float* bq = (const float*)d_in[2];
  const float* wk = (const float*)d_in[3];
  const float* bk = (const float*)d_in[4];
  const float* wv = (const float*)d_in[5];
  const float* bv = (const float*)d_in[6];
  const float* wo = (const float*)d_in[7];
  const float* bo = (const float*)d_in[8];

  // workspace layout (bf16): x, 4 weights, Q, K, V^T, ctx  (~48 MB total)
  u16* xb  = (u16*)d_ws;
  u16* wqb = xb  + (size_t)Nc * Dc;
  u16* wkb = wqb + (size_t)Dc * Dc;
  u16* wvb = wkb + (size_t)Dc * Dc;
  u16* wob = wvb + (size_t)Dc * Dc;
  u16* qb  = wob + (size_t)Dc * Dc;
  u16* kb  = qb  + (size_t)Nc * Dc;
  u16* vt  = kb  + (size_t)Nc * Dc;
  u16* ctx = vt  + (size_t)Nc * Dc;

  const int nx = Nc * Dc, nw = Dc * Dc;
  mhsa_cvt_bf16<<<(nx / 4 + 255) / 256, 256, 0, stream>>>(x,  xb,  nx);
  mhsa_cvt_bf16<<<(nw / 4 + 255) / 256, 256, 0, stream>>>(wq, wqb, nw);
  mhsa_cvt_bf16<<<(nw / 4 + 255) / 256, 256, 0, stream>>>(wk, wkb, nw);
  mhsa_cvt_bf16<<<(nw / 4 + 255) / 256, 256, 0, stream>>>(wv, wvb, nw);
  mhsa_cvt_bf16<<<(nw / 4 + 255) / 256, 256, 0, stream>>>(wo, wob, nw);

  // GEMM: 32 m-blocks x 16 n-blocks = 512 blocks of 256 threads
  mhsa_gemm_bf16<<<512, 256, 0, stream>>>(xb, wqb, bq, qb, 0);
  mhsa_gemm_bf16<<<512, 256, 0, stream>>>(xb, wkb, bk, kb, 0);
  mhsa_gemm_bf16<<<512, 256, 0, stream>>>(xb, wvb, bv, vt, 1);

  // attention: 4096 wave-tiles / 8 waves per block = 512 blocks
  mhsa_flash_attn<<<512, 256, 0, stream>>>(qb, kb, vt, ctx);

  mhsa_gemm_bf16<<<512, 256, 0, stream>>>(ctx, wob, bo, d_out, 2);
}
